// MultiModalHAN_74998718923078
// MI455X (gfx1250) — matrix operand order repaired and checked by reading, not compiled, not run on hardware
//
#include <hip/hip_runtime.h>

typedef __attribute__((ext_vector_type(16))) _Float16 v16h;
typedef __attribute__((ext_vector_type(8)))  _Float16 v8h;
typedef __attribute__((ext_vector_type(4)))  _Float16 v4h;
typedef __attribute__((ext_vector_type(2)))  _Float16 v2h;
typedef __attribute__((ext_vector_type(8)))  float    v8f;

#define NN    8192
#define LL    32
#define DD    300
#define DP    320      // DD padded to mult of 64
#define PP    256
#define GG    64
#define EE    131072
#define GATE  1200
#define GATEP 1216     // GATE padded to mult of 64
#define KP0   608      // (300+300) padded to mult of 32
#define KP1   928      // (600+300) padded to mult of 32

// ---------------------------------------------------------------- utilities
__global__ void k_fill(float* p, size_t n, float v) {
  size_t i = (size_t)blockIdx.x * 256 + threadIdx.x;
  if (i < n) p[i] = v;
}

__global__ void k_cvt16(const float* s, _Float16* d, size_t n) {
  size_t i = (size_t)blockIdx.x * 256 + threadIdx.x;
  if (i < n) d[i] = (_Float16)s[i];
}

// pe16[k][DP]: pad cols 300..319 with zero (no transpose)
__global__ void k_pe(const float* src, _Float16* dst) {
  int i = blockIdx.x * 256 + threadIdx.x;
  if (i >= PP * DP) return;
  int k = i / DP, n = i % DP;
  dst[i] = (_Float16)((n < DD) ? src[k * DD + n] : 0.f);
}

// GAT/att weight: src [300x300] row-major -> dst [320][320] = src^T, zero-padded
__global__ void k_trp(const float* src, _Float16* dst) {
  int i = blockIdx.x * 256 + threadIdx.x;
  if (i >= DP * DP) return;
  int k = i / DP, n = i % DP;
  dst[i] = (_Float16)((k < DD && n < DD) ? src[n * DD + k] : 0.f);
}

// Wcat [Kp][1216] = [Wih^T ; Whh^T], zero-padded rows/cols
__global__ void k_wcat(const float* Wih, const float* Whh, _Float16* dst,
                       int in_dim, int Kp) {
  int i = blockIdx.x * 256 + threadIdx.x;
  if (i >= Kp * GATEP) return;
  int k = i / GATEP, n = i % GATEP;
  float v = 0.f;
  if (n < GATE) {
    if (k < in_dim)            v = Wih[(size_t)n * in_dim + k];
    else if (k < in_dim + DD)  v = Whh[(size_t)n * DD + (k - in_dim)];
  }
  dst[i] = (_Float16)v;
}

__global__ void k_addb(const float* a, const float* b, float* o) {
  int i = blockIdx.x * 256 + threadIdx.x;
  if (i < GATEP) o[i] = (i < GATE) ? (a[i] + b[i]) : 0.f;
}

__global__ void k_padb(const float* src, float* dst) {
  int i = blockIdx.x * 256 + threadIdx.x;
  if (i < DP) dst[i] = (i < DD) ? src[i] : 0.f;
}

__device__ __forceinline__ void atomicMaxF(float* addr, float v) {
  if (v >= 0.f) atomicMax((int*)addr, __float_as_int(v));
  else          atomicMin((unsigned int*)addr, __float_as_uint(v));
}

// CDNA5 async global->LDS copy (16B per lane), tracked by ASYNCcnt.
__device__ __forceinline__ void async_copy16(const _Float16* g, _Float16* l) {
  unsigned lds = (unsigned)(uintptr_t)l;  // low 32 bits of LDS-aperture addr
  asm volatile("global_load_async_to_lds_b128 %0, %1, off"
               :: "v"(lds), "v"(g) : "memory");
}
__device__ __forceinline__ void wait_async() {
  asm volatile("s_wait_asynccnt 0" ::: "memory");
}

// ---------------------------------------------------------------- WMMA GEMM
// C[M,Np](f32) = A[M,Kp](f16) * B[Kp,Np](f16) + bias[n]
// Caller guarantees: M%64==0, Np%64==0, Kp%32==0, A pad cols zero, B pad rows
// zero -> no bounds checks. 256 thr = 8 waves -> 64x64 tile; each wave owns a
// 16x32 strip (2 WMMAs sharing its A fragment). Double-buffered LDS:
//   A tile staged via global_load_async_to_lds_b128 (ASYNCcnt),
//   B tile transposed through VGPRs ([n][k] in LDS -> contiguous b128 frags),
// with next-chunk loads issued before current-chunk WMMAs.
__global__ __launch_bounds__(256) void gemm_f16(const _Float16* __restrict__ A,
                                                const _Float16* __restrict__ B,
                                                float* __restrict__ C,
                                                const float* __restrict__ bias,
                                                int Np, int Kp) {
  __shared__ __align__(32) _Float16 sA[2][64 * 32];   // [m][k]
  __shared__ __align__(32) _Float16 sBt[2][64 * 32];  // [n][k]
  const int bm = blockIdx.y * 64;
  const int bn = blockIdx.x * 64;
  const int tid  = threadIdx.x;
  const int wave = tid >> 5, lane = tid & 31;
  const int wm  = (wave >> 1) * 16;     // 4 M-strips
  const int wnb = (wave & 1) * 32;      // 2 N-strips of 32
  const int l15 = lane & 15;
  const int hi  = lane >> 4;
  const int mrow = wm + l15;
  const int ahi  = hi * 16;             // A K-chunk base {0,16}
  const int bhi  = hi * 16;             // B K-chunk base {0,16}
  // staging coords: A 64x32, 8 halves (16B) per thread (async copy)
  const int ar = tid >> 2, ac = (tid & 3) * 8;
  // staging coords: B 32x64, k-pair x 4n per thread (4x packed b32 LDS stores)
  const int bk = (tid >> 4) * 2, bnc = (tid & 15) * 4;

  const _Float16* aSrc = A + (size_t)(bm + ar) * Kp + ac;
  const _Float16* bSrc = B + (size_t)bk * Np + bn + bnc;
  const size_t bStep = (size_t)32 * Np;
  const int nk = Kp >> 5;

  v8f acc0 = {}, acc1 = {};

  // prologue: stage chunk 0 into buffer 0
  async_copy16(aSrc, &sA[0][ar * 32 + ac]);
  v4h blo = *(const v4h*)bSrc;
  v4h bhi4 = *(const v4h*)(bSrc + Np);
#pragma unroll
  for (int j = 0; j < 4; ++j) {
    v2h p = {blo[j], bhi4[j]};
    *(v2h*)(&sBt[0][(bnc + j) * 32 + bk]) = p;
  }
  wait_async();
  __syncthreads();

  for (int i = 0; i < nk; ++i) {
    const int cur = i & 1, nxt = cur ^ 1;
    v4h nlo, nhi;
    if (i + 1 < nk) {  // issue next-chunk loads before computing
      async_copy16(aSrc + (size_t)(i + 1) * 32, &sA[nxt][ar * 32 + ac]);
      const _Float16* bs = bSrc + (size_t)(i + 1) * bStep;
      nlo = *(const v4h*)bs;
      nhi = *(const v4h*)(bs + Np);
    }
    // A frag: lane m / m+16 hold K 0-15 / 16-31
    v8h alo = *(const v8h*)(&sA[cur][mrow * 32 + ahi]);
    v8h ah8 = *(const v8h*)(&sA[cur][mrow * 32 + ahi + 8]);
    v16h a  = __builtin_shufflevector(alo, ah8, 0, 1, 2, 3, 4, 5, 6, 7,
                                      8, 9, 10, 11, 12, 13, 14, 15);
    // B frag: lane n / n+16 hold K 0-15 / 16-31 (contiguous in sBt)
    v16h b0 = *(const v16h*)(&sBt[cur][(wnb + l15) * 32 + bhi]);
    v16h b1 = *(const v16h*)(&sBt[cur][(wnb + 16 + l15) * 32 + bhi]);
    acc0 = __builtin_amdgcn_wmma_f32_16x16x32_f16(false, a, false, b0,
                                                  (short)0, acc0, false, false);
    acc1 = __builtin_amdgcn_wmma_f32_16x16x32_f16(false, a, false, b1,
                                                  (short)0, acc1, false, false);
    if (i + 1 < nk) {
#pragma unroll
      for (int j = 0; j < 4; ++j) {
        v2h p = {nlo[j], nhi[j]};
        *(v2h*)(&sBt[nxt][(bnc + j) * 32 + bk]) = p;
      }
      wait_async();
    }
    __syncthreads();
  }

  const int n0 = bn + wnb + l15;
  const float bs0 = bias ? bias[n0] : 0.f;
  const float bs1 = bias ? bias[n0 + 16] : 0.f;
#pragma unroll
  for (int r = 0; r < 8; ++r) {
    int m = bm + wm + r + hi * 8;       // D: vgpr r -> M = r + 8*(lane>=16)
    C[(size_t)m * Np + n0]      = acc0[r] + bs0;
    C[(size_t)m * Np + n0 + 16] = acc1[r] + bs1;
  }
}

// ---------------------------------------------------------------- LSTM
__global__ void k_xcat_l0(const int* tokens, const int* lengths, const float* wemb,
                          const float* h, _Float16* xcat, int t, int back) {
  int col = blockIdx.x * 256 + threadIdx.x;
  if (col >= KP0) return;
  int n = blockIdx.y;
  float v = 0.f;
  if (col < DD) {
    int len = lengths[n];
    int pos = back ? ((t < len) ? (len - 1 - t) : t) : t;
    int tok = tokens[n * LL + pos];
    v = wemb[(size_t)tok * DD + col];
  } else if (col < 2 * DD) {
    v = h[(size_t)n * DD + (col - DD)];
  }
  xcat[(size_t)n * KP0 + col] = (_Float16)v;
}

__global__ void k_xcat_l1(const int* lengths, const _Float16* o1, const float* h,
                          _Float16* xcat, int t, int back) {
  int col = blockIdx.x * 256 + threadIdx.x;
  if (col >= KP1) return;
  int n = blockIdx.y;
  _Float16 v = (_Float16)0.f;
  if (col < 600) {
    int len = lengths[n];
    int pos = back ? ((t < len) ? (len - 1 - t) : t) : t;
    v = o1[((size_t)n * LL + pos) * 600 + col];
  } else if (col < 900) {
    v = (_Float16)h[(size_t)n * DD + (col - 600)];
  }
  xcat[(size_t)n * KP1 + col] = v;
}

__global__ void k_lstm_cell(const float* gates, const int* lengths, float* h, float* c,
                            _Float16* seq, int t, int back, int dirOff) {
  int i = blockIdx.x * 256 + threadIdx.x;
  if (i >= NN * DD) return;
  int n = i / DD, d = i % DD;
  int len = lengths[n];
  const float* g = gates + (size_t)n * GATEP;
  float gi = g[d], gf = g[DD + d], gg = g[2 * DD + d], go = g[3 * DD + d];
  float si = 1.f / (1.f + __expf(-gi));
  float sf = 1.f / (1.f + __expf(-gf));
  float so = 1.f / (1.f + __expf(-go));
  float hi = h[i], ci = c[i];
  float cn = sf * ci + si * tanhf(gg);
  float hn = so * tanhf(cn);
  if (t < len) { hi = hn; ci = cn; }
  h[i] = hi; c[i] = ci;
  if (seq) {
    int pos = back ? ((t < len) ? (len - 1 - t) : t) : t;  // un-reverse bwd dir
    seq[((size_t)n * LL + pos) * 600 + dirOff + d] = (_Float16)hi;
  }
}

// ---------------------------------------------------------------- encoder mux
__global__ void k_enc(const int* ntype, const float* denc, const float* cenc,
                      const float* h0f, const float* h0b, const float* h1f,
                      const float* h1b, _Float16* x16) {
  int i = blockIdx.x * 256 + threadIdx.x;
  if (i >= NN * DP) return;
  int n = i / DP, d = i % DP;
  float v = 0.f;
  if (d < DD) {
    int tp = ntype[n];
    size_t j = (size_t)n * DD + d;
    if (tp == 1)      v = denc[i];
    else if (tp == 3) v = cenc[i];
    else              v = 0.25f * (h0f[j] + h0b[j] + h1f[j] + h1b[j]);
  }
  x16[i] = (_Float16)v;
}

// ---------------------------------------------------------------- GAT
__global__ void k_dot2(const float* xp, const float* a_s, const float* a_d,
                       float* als, float* ald) {
  int node = blockIdx.x * 8 + (threadIdx.x >> 5);
  int lane = threadIdx.x & 31;
  if (node >= NN) return;
  float s0 = 0.f, s1 = 0.f;
  const float* row = xp + (size_t)node * DP;
  for (int d = lane; d < DD; d += 32) { float x = row[d]; s0 += x * a_s[d]; s1 += x * a_d[d]; }
  for (int off = 16; off > 0; off >>= 1) { s0 += __shfl_down(s0, off, 32); s1 += __shfl_down(s1, off, 32); }
  if (lane == 0) { als[node] = s0; ald[node] = s1; }
}

__device__ __forceinline__ void edge_sd(const int* ei, int idx, int& s, int& d) {
  if (idx < EE) { s = ei[idx]; d = ei[EE + idx]; } else { s = d = idx - EE; }  // self loops
}

__global__ void k_emax(const int* ei, const float* als, const float* ald, float* m) {
  int idx = blockIdx.x * 256 + threadIdx.x;
  if (idx >= EE + NN) return;
  int s, d; edge_sd(ei, idx, s, d);
  float e = als[s] + ald[d];
  e = e > 0.f ? e : 0.2f * e;
  atomicMaxF(&m[d], e);
}

__global__ void k_eden(const int* ei, const float* als, const float* ald,
                       const float* m, float* den) {
  int idx = blockIdx.x * 256 + threadIdx.x;
  if (idx >= EE + NN) return;
  int s, d; edge_sd(ei, idx, s, d);
  float e = als[s] + ald[d];
  e = e > 0.f ? e : 0.2f * e;
  atomicAdd(&den[d], __expf(e - m[d]));
}

__global__ void k_eaggr(const int* ei, const float* als, const float* ald,
                        const float* m, const float* den, const float* xp,
                        float* z, int et) {
  int idx  = blockIdx.x * 8 + (threadIdx.x >> 5);
  int lane = threadIdx.x & 31;
  if (idx >= EE + NN) return;
  int s, d; edge_sd(ei, idx, s, d);
  float e = als[s] + ald[d];
  e = e > 0.f ? e : 0.2f * e;
  float alpha = __expf(e - m[d]) / den[d];
  float* zrow = z + ((size_t)d * 3 + et) * DP;
  const float* xr = xp + (size_t)s * DP;
  for (int k = lane; k < DD; k += 32) atomicAdd(&zrow[k], alpha * xr[k]);
}

__global__ void k_zb(float* z, const float* b1, const float* b2, const float* b3,
                     _Float16* z16) {
  int i = blockIdx.x * 256 + threadIdx.x;
  if (i >= NN * 3 * DP) return;
  int rem = i % (3 * DP);
  int et = rem / DP, d = rem % DP;
  float v = 0.f;
  if (d < DD) {
    const float* b = (et == 0) ? b1 : (et == 1) ? b2 : b3;
    v = z[i] + b[d];
  }
  z[i] = v;
  z16[i] = (_Float16)v;
}

// ---------------------------------------------------------------- HAN combine
__global__ void k_comb(const float* T, const float* attv, const float* z,
                       float* xout, _Float16* x16) {
  int node = blockIdx.x * 8 + (threadIdx.x >> 5);
  int lane = threadIdx.x & 31;
  if (node >= NN) return;
  float w[3];
#pragma unroll
  for (int e = 0; e < 3; ++e) {
    float s = 0.f;
    const float* row = T + ((size_t)node * 3 + e) * DP;
    for (int d = lane; d < DD; d += 32) s += tanhf(row[d]) * attv[d];
    for (int off = 16; off > 0; off >>= 1) s += __shfl_down(s, off, 32);
    w[e] = __shfl(s, 0, 32);
  }
  float mx = fmaxf(w[0], fmaxf(w[1], w[2]));
  float e0 = __expf(w[0] - mx), e1 = __expf(w[1] - mx), e2 = __expf(w[2] - mx);
  float inv = 1.f / (e0 + e1 + e2);
  float b0 = e0 * inv, b1 = e1 * inv, b2 = e2 * inv;
  const float* zr = z + (size_t)node * 3 * DP;
  for (int d = lane; d < DD; d += 32) {
    float v = b0 * zr[d] + b1 * zr[DP + d] + b2 * zr[2 * DP + d];
    xout[(size_t)node * DD + d] = v;
    x16[(size_t)node * DP + d]  = (_Float16)v;   // pad cols stay zero from k_enc
  }
}

// ---------------------------------------------------------------- pooling
__global__ void k_pooladd(const int* batch, const float* x, float* s, float* cnt) {
  int node = blockIdx.x * 8 + (threadIdx.x >> 5);
  int lane = threadIdx.x & 31;
  if (node >= NN) return;
  int g = batch[node];
  for (int d = lane; d < DD; d += 32) atomicAdd(&s[(size_t)g * DD + d], x[(size_t)node * DD + d]);
  if (lane == 0) atomicAdd(&cnt[g], 1.0f);
}

__global__ void k_pooldiv(const float* s, const float* cnt, float* out) {
  int i = blockIdx.x * 256 + threadIdx.x;
  if (i >= GG * DD) return;
  out[i] = s[i] / fmaxf(cnt[i / DD], 1.f);
}

// ---------------------------------------------------------------- host
static inline void launch_gemm(const _Float16* A, const _Float16* B, float* C,
                               const float* bias, int M, int Np, int Kp,
                               hipStream_t s) {
  dim3 grid(Np / 64, M / 64);
  gemm_f16<<<grid, 256, 0, s>>>(A, B, C, bias, Np, Kp);
}

extern "C" void kernel_launch(void* const* d_in, const int* in_sizes, int n_in,
                              void* d_out, int out_size, void* d_ws, size_t ws_size,
                              hipStream_t stream) {
  const int*   tokens   = (const int*)d_in[0];
  const int*   lengths  = (const int*)d_in[1];
  const int*   ntype    = (const int*)d_in[2];
  const int*   edges[3] = {(const int*)d_in[3], (const int*)d_in[4], (const int*)d_in[5]};
  const int*   batch    = (const int*)d_in[6];
  const float* prof     = (const float*)d_in[7];
  const float* wemb     = (const float*)d_in[8];
  const float* dpe      = (const float*)d_in[9];
  const float* cpe      = (const float*)d_in[10];

  // Param flatten order: insertion vs jax-sorted dicts — detect via in_sizes[31]
  // insertion: idx31 = layer1.g2.W (90000); sorted: idx31 = layer1.g1.a_s (300)
  bool sorted = (n_in > 31) && (in_sizes[31] == 300);
  int lstmIdx[4];                 // runs: l0f, l0b, l1f, l1b
  if (!sorted) { lstmIdx[0] = 11; lstmIdx[1] = 15; lstmIdx[2] = 19; lstmIdx[3] = 23; }
  else         { lstmIdx[0] = 15; lstmIdx[1] = 11; lstmIdx[2] = 23; lstmIdx[3] = 19; }

  // ---- workspace carve (all 256B aligned)
  char* w = (char*)d_ws;
  auto carve = [&](size_t bytes) -> char* {
    char* p = w; w += (bytes + 255) & ~(size_t)255; return p;
  };
  _Float16* o1     = (_Float16*)carve((size_t)NN * LL * 600 * 2);
  _Float16* xcat   = (_Float16*)carve((size_t)NN * KP1 * 2);
  float*    gates  = (float*)carve((size_t)NN * GATEP * 4);
  float*    hT[4];
  for (int r = 0; r < 4; ++r) hT[r] = (float*)carve((size_t)NN * DD * 4);
  float*    cst    = (float*)carve((size_t)NN * DD * 4);
  _Float16* Wcat   = (_Float16*)carve((size_t)KP1 * GATEP * 2);
  float*    bcat   = (float*)carve((size_t)GATEP * 4);
  _Float16* Wg     = (_Float16*)carve((size_t)DP * DP * 2);
  _Float16* Wa     = (_Float16*)carve((size_t)DP * DP * 2);
  _Float16* pe16   = (_Float16*)carve((size_t)PP * DP * 2);
  _Float16* prof16 = (_Float16*)carve((size_t)NN * PP * 2);
  float*    denc   = (float*)carve((size_t)NN * DP * 4);
  float*    cenc   = (float*)carve((size_t)NN * DP * 4);
  _Float16* xin16  = (_Float16*)carve((size_t)NN * DP * 2);
  float*    xp     = (float*)carve((size_t)NN * DP * 4);
  float*    als    = (float*)carve((size_t)NN * 4);
  float*    ald    = (float*)carve((size_t)NN * 4);
  float*    mbuf   = (float*)carve((size_t)NN * 4);
  float*    dbuf   = (float*)carve((size_t)NN * 4);
  float*    z      = (float*)carve((size_t)NN * 3 * DP * 4);
  _Float16* z16    = (_Float16*)carve((size_t)NN * 3 * DP * 2);
  float*    Tbuf   = (float*)carve((size_t)NN * 3 * DP * 4);
  float*    xout   = (float*)carve((size_t)NN * DD * 4);
  float*    battP  = (float*)carve((size_t)DP * 4);
  float*    pools  = (float*)carve((size_t)GG * DD * 4);
  float*    poolc  = (float*)carve((size_t)GG * 4);

#define GRID1(n) dim3((unsigned)(((n) + 255) / 256)), dim3(256), 0, stream
#define GRIDW(n) dim3((unsigned)(((n) + 7) / 8)), dim3(256), 0, stream

  // ---- profile encoders: [N,256]f16 @ [256,320]f16
  k_cvt16<<<GRID1((size_t)NN * PP)>>>(prof, prof16, (size_t)NN * PP);
  k_pe<<<GRID1(PP * DP)>>>(dpe, pe16);
  launch_gemm(prof16, pe16, denc, nullptr, NN, DP, PP, stream);
  k_pe<<<GRID1(PP * DP)>>>(cpe, pe16);
  launch_gemm(prof16, pe16, cenc, nullptr, NN, DP, PP, stream);

  // ---- BiLSTM (2 layers x 2 dirs): per-step fused GEMM [x_t|h]@[Wih^T;Whh^T]
  for (int r = 0; r < 4; ++r) {
    const int layer = r >> 1, back = r & 1;
    const int in_dim = layer ? 600 : 300;
    const int Kp = layer ? KP1 : KP0;
    const float* Wih = (const float*)d_in[lstmIdx[r]];
    const float* Whh = (const float*)d_in[lstmIdx[r] + 1];
    const float* bih = (const float*)d_in[lstmIdx[r] + 2];
    const float* bhh = (const float*)d_in[lstmIdx[r] + 3];
    k_wcat<<<GRID1(Kp * GATEP)>>>(Wih, Whh, Wcat, in_dim, Kp);
    k_addb<<<GRID1(GATEP)>>>(bih, bhh, bcat);
    k_fill<<<GRID1((size_t)NN * DD)>>>(hT[r], (size_t)NN * DD, 0.f);
    k_fill<<<GRID1((size_t)NN * DD)>>>(cst, (size_t)NN * DD, 0.f);
    dim3 xg((Kp + 255) / 256, NN);
    for (int t = 0; t < LL; ++t) {
      if (layer == 0)
        k_xcat_l0<<<xg, 256, 0, stream>>>(tokens, lengths, wemb, hT[r], xcat, t, back);
      else
        k_xcat_l1<<<xg, 256, 0, stream>>>(lengths, o1, hT[r], xcat, t, back);
      launch_gemm(xcat, Wcat, gates, bcat, NN, GATEP, Kp, stream);
      k_lstm_cell<<<GRID1(NN * DD)>>>(gates, lengths, hT[r], cst,
                                      layer == 0 ? o1 : (_Float16*)nullptr,
                                      t, back, back ? DD : 0);
    }
  }

  // ---- encoder mux -> xin16 (pad cols zeroed)
  k_enc<<<GRID1(NN * DP)>>>(ntype, denc, cenc, hT[0], hT[1], hT[2], hT[3], xin16);

  // ---- 2x HAN layer (3x GAT + semantic attention)
  for (int Ly = 0; Ly < 2; ++Ly) {
    const int pbase = 27 + Ly * 15;
    int gatIdx[3], attWi, attBi, attVi;
    if (!sorted) {
      for (int e = 0; e < 3; ++e) gatIdx[e] = pbase + e * 4;
      attWi = pbase + 12; attBi = pbase + 13; attVi = pbase + 14;
    } else {
      attWi = pbase; attBi = pbase + 1; attVi = pbase + 2;
      for (int e = 0; e < 3; ++e) gatIdx[e] = pbase + 3 + e * 4;
    }
    k_fill<<<GRID1((size_t)NN * 3 * DP)>>>(z, (size_t)NN * 3 * DP, 0.f);
    const int tot = EE + NN;
    for (int et = 0; et < 3; ++et) {
      const float* Wp  = (const float*)d_in[gatIdx[et]];
      const float* a_s = (const float*)d_in[gatIdx[et] + 1];
      const float* a_d = (const float*)d_in[gatIdx[et] + 2];
      k_trp<<<GRID1(DP * DP)>>>(Wp, Wg);
      launch_gemm(xin16, Wg, xp, nullptr, NN, DP, DP, stream);
      k_dot2<<<GRIDW(NN)>>>(xp, a_s, a_d, als, ald);
      k_fill<<<GRID1((size_t)NN)>>>(mbuf, (size_t)NN, -3.0e38f);
      k_fill<<<GRID1((size_t)NN)>>>(dbuf, (size_t)NN, 0.f);
      k_emax<<<GRID1(tot)>>>(edges[et], als, ald, mbuf);
      k_eden<<<GRID1(tot)>>>(edges[et], als, ald, mbuf, dbuf);
      k_eaggr<<<GRIDW(tot)>>>(edges[et], als, ald, mbuf, dbuf, xp, z, et);
    }
    k_zb<<<GRID1(NN * 3 * DP)>>>(z, (const float*)d_in[gatIdx[0] + 3],
                                 (const float*)d_in[gatIdx[1] + 3],
                                 (const float*)d_in[gatIdx[2] + 3], z16);
    k_trp<<<GRID1(DP * DP)>>>((const float*)d_in[attWi], Wa);
    k_padb<<<GRID1(DP)>>>((const float*)d_in[attBi], battP);
    launch_gemm(z16, Wa, Tbuf, battP, NN * 3, DP, DP, stream);
    k_comb<<<GRIDW(NN)>>>(Tbuf, (const float*)d_in[attVi], z, xout, xin16);
  }

  // ---- per-graph mean pooling
  k_fill<<<GRID1((size_t)GG * DD)>>>(pools, (size_t)GG * DD, 0.f);
  k_fill<<<GRID1((size_t)GG)>>>(poolc, (size_t)GG, 0.f);
  k_pooladd<<<GRIDW(NN)>>>(batch, xout, pools, poolc);
  k_pooldiv<<<GRID1(GG * DD)>>>(pools, poolc, (float*)d_out);

  (void)out_size; (void)ws_size;
#undef GRID1
#undef GRIDW
}
